// GraphConvolutionalModule_29360396436005
// MI455X (gfx1250) — matrix operand order repaired and checked by reading, not compiled, not run on hardware
//
#include <hip/hip_runtime.h>

#define N_NODES 10000
#define N_EDGES 160000
#define BATCH   4
#define D       128
#define ND      (N_NODES * D)

typedef __attribute__((ext_vector_type(16))) _Float16 v16h;
typedef __attribute__((ext_vector_type(8)))  float    v8f;

struct WPtrs { const float* w[16]; };

// ---------------------------------------------------------------------------
// Weight prep: convert 16 [128x128] f32 row-major (k,n) matrices to f16
// transposed (n,k) so the WMMA B operand reads two runs of 8 contiguous halfs per lane.
// ---------------------------------------------------------------------------
__global__ void prep_weights_kernel(WPtrs wp, _Float16* __restrict__ Wt) {
  int mat = blockIdx.y;
  int i = blockIdx.x * blockDim.x + threadIdx.x;   // 0..16383
  int k = i >> 7, n = i & 127;
  Wt[mat * 16384 + n * 128 + k] = (_Float16)wp.w[mat][k * 128 + n];
}

__global__ void zero_pool_kernel(float* pool) {
  int i = blockIdx.x * blockDim.x + threadIdx.x;
  if (i < BATCH * 2 * D) pool[i] = 0.f;
}

// ---------------------------------------------------------------------------
// Fused q/k/v/skip GEMM: Y[w] = X @ W[w] + b[w] for w = 0..3, one wave per
// 16x16 output tile. The A operand (X tile) is loaded & converted ONCE per
// K-step and shared by 4 independent v_wmma_f32_16x16x32_f16 issues (one per
// weight matrix), so the matrix pipe sees 16 WMMAs per wave with no
// accumulator hazards between consecutive issues.
// A layout (16-bit A 16x32, ISA 7.12.2): lanes 0-15 hold K {0..7,16..23},
// lanes 16-31 hold K {8..15,24..31} for row M = lane%16.
// B layout: lane = column N, lanes 0-15 K {0..7,16..23}, lanes 16-31 K {8..15,24..31}.
// C/D: VGPR r -> row M = r (+8 for lanes 16-31), col N = lane%16.
// ---------------------------------------------------------------------------
__global__ void gemm_qkvs_kernel(const float* __restrict__ X,
                                 const _Float16* __restrict__ Wt4,
                                 const float* __restrict__ pbq,
                                 const float* __restrict__ pbk,
                                 const float* __restrict__ pbv,
                                 const float* __restrict__ pbs,
                                 float* __restrict__ Y4) {
  const int lane = threadIdx.x;
  const int m0 = blockIdx.x << 4;
  const int n0 = blockIdx.y << 4;

  const int half16 = lane >> 4;    // 0: lanes 0-15, 1: lanes 16-31
  const int l      = lane & 15;
  const int rowA = m0 + l;
  const int colB = n0 + l;

  v8f c0 = {}, c1 = {}, c2 = {}, c3 = {};
#pragma unroll
  for (int kb = 0; kb < 128; kb += 32) {
    // ---- A tile (shared by all 4 weights) ----
    v16h a;
    const int kA0 = kb + (half16 << 3);            // +0 / +8
    const float* xr = X + rowA * 128 + kA0;
#pragma unroll
    for (int i = 0; i < 8; ++i) {
      a[i]     = (_Float16)xr[i];                  // K = kA0 + i
      a[i + 8] = (_Float16)xr[i + 16];             // K = kA0 + 16 + i
    }
    // ---- B tiles for the 4 weight matrices ----
    const int kB0 = kb + (half16 << 3);            // +0 / +8
    const _Float16* wr = Wt4 + colB * 128 + kB0;
    v16h b0, b1, b2, b3;
#pragma unroll
    for (int i = 0; i < 16; ++i) {
      const int kk = (i < 8) ? i : i + 8;
      b0[i] = wr[kk];
      b1[i] = wr[16384 + kk];
      b2[i] = wr[32768 + kk];
      b3[i] = wr[49152 + kk];
    }
    c0 = __builtin_amdgcn_wmma_f32_16x16x32_f16(false, a, false, b0, (short)0, c0, false, false);
    c1 = __builtin_amdgcn_wmma_f32_16x16x32_f16(false, a, false, b1, (short)0, c1, false, false);
    c2 = __builtin_amdgcn_wmma_f32_16x16x32_f16(false, a, false, b2, (short)0, c2, false, false);
    c3 = __builtin_amdgcn_wmma_f32_16x16x32_f16(false, a, false, b3, (short)0, c3, false, false);
  }
  const float b0s = pbq[colB], b1s = pbk[colB], b2s = pbv[colB], b3s = pbs[colB];
#pragma unroll
  for (int r = 0; r < 8; ++r) {
    int row = m0 + r + (half16 << 3);
    Y4[(size_t)0 * ND + row * 128 + colB] = c0[r] + b0s;
    Y4[(size_t)1 * ND + row * 128 + colB] = c1[r] + b1s;
    Y4[(size_t)2 * ND + row * 128 + colB] = c2[r] + b2s;
    Y4[(size_t)3 * ND + row * 128 + colB] = c3[r] + b3s;
  }
}

// ---------------------------------------------------------------------------
// Segment-op scratch init: AGG = 0, M = -inf, DEN = 0.
// ---------------------------------------------------------------------------
__global__ void init_seg_kernel(float* __restrict__ AGG, float* __restrict__ M,
                                float* __restrict__ DEN) {
  int i = blockIdx.x * blockDim.x + threadIdx.x;
  if (i < ND) AGG[i] = 0.f;
  if (i < N_NODES) { M[i] = -__builtin_inff(); DEN[i] = 0.f; }
}

// float atomic max via signed/unsigned int ordering trick (native atomics).
__device__ __forceinline__ void atomicMaxF(float* addr, float v) {
  if (v >= 0.f) atomicMax((int*)addr, __float_as_int(v));
  else          atomicMin((unsigned int*)addr, __float_as_uint(v));
}

// ---------------------------------------------------------------------------
// Edge pass 1: score[e] = dot(q[dst], k[src] + t*We)/sqrt(128); segment max.
// One wave per edge; 4 floats per lane; shfl_xor tree reduction.
// ---------------------------------------------------------------------------
__global__ void edge_score_kernel(const float* __restrict__ Q,
                                  const float* __restrict__ K,
                                  const float* __restrict__ We,
                                  const float* __restrict__ ea,
                                  const int* __restrict__ src,
                                  const int* __restrict__ dst,
                                  float* __restrict__ score,
                                  float* __restrict__ M) {
  int gid = blockIdx.x * blockDim.x + threadIdx.x;
  int e = gid >> 5;
  if (e >= N_EDGES) return;
  int lane = threadIdx.x & 31;
  int s = src[e], d = dst[e];
  float t = ea[e];
  const float4 qv = *(const float4*)(Q + (size_t)d * 128 + lane * 4);
  const float4 kv = *(const float4*)(K + (size_t)s * 128 + lane * 4);
  const float4 wv = *(const float4*)(We + lane * 4);
  float acc = qv.x * (kv.x + t * wv.x) + qv.y * (kv.y + t * wv.y) +
              qv.z * (kv.z + t * wv.z) + qv.w * (kv.w + t * wv.w);
#pragma unroll
  for (int m = 16; m >= 1; m >>= 1) acc += __shfl_xor(acc, m, 32);
  if (lane == 0) {
    float sc = acc * 0.08838834764831845f;   // 1/sqrt(128)
    score[e] = sc;
    atomicMaxF(M + d, sc);
  }
}

// Edge pass 2: ex = exp(score - m[dst]); segment sum into DEN. (in-place)
__global__ void edge_den_kernel(const int* __restrict__ dst,
                                const float* __restrict__ M,
                                float* __restrict__ score,
                                float* __restrict__ DEN) {
  int e = blockIdx.x * blockDim.x + threadIdx.x;
  if (e >= N_EDGES) return;
  int d = dst[e];
  float ex = __expf(score[e] - M[d]);
  score[e] = ex;
  atomicAdd(DEN + d, ex);
}

// Edge pass 3: AGG[dst] += alpha * (v[src] + t*We). One wave per edge.
__global__ void edge_agg_kernel(const float* __restrict__ V,
                                const float* __restrict__ We,
                                const float* __restrict__ ea,
                                const int* __restrict__ src,
                                const int* __restrict__ dst,
                                const float* __restrict__ exs,
                                const float* __restrict__ DEN,
                                float* __restrict__ AGG) {
  int gid = blockIdx.x * blockDim.x + threadIdx.x;
  int e = gid >> 5;
  if (e >= N_EDGES) return;
  int lane = threadIdx.x & 31;
  int s = src[e], d = dst[e];
  float alpha = exs[e] / fmaxf(DEN[d], 1e-16f);
  float t = ea[e];
  const float4 vv = *(const float4*)(V + (size_t)s * 128 + lane * 4);
  const float4 wv = *(const float4*)(We + lane * 4);
  float* out = AGG + (size_t)d * 128 + lane * 4;
  atomicAdd(out + 0, alpha * (vv.x + t * wv.x));
  atomicAdd(out + 1, alpha * (vv.y + t * wv.y));
  atomicAdd(out + 2, alpha * (vv.z + t * wv.z));
  atomicAdd(out + 3, alpha * (vv.w + t * wv.w));
}

// Layer-1 epilogue: H = relu(AGG + skip).
__global__ void finalize_relu_kernel(const float* __restrict__ AGG,
                                     const float* __restrict__ S,
                                     float* __restrict__ H) {
  int i = blockIdx.x * blockDim.x + threadIdx.x;
  if (i < ND) H[i] = fmaxf(AGG[i] + S[i], 0.f);
}

// Layer-2 epilogue + mean-pool partial sums: pool[j] += sum_rows(AGG + S).
__global__ void pool_kernel(const float* __restrict__ AGG,
                            const float* __restrict__ S,
                            float* __restrict__ pool) {
  int j = threadIdx.x;                 // 0..127 (column)
  float acc = 0.f;
  for (int r = blockIdx.x; r < N_NODES; r += gridDim.x)
    acc += AGG[r * 128 + j] + S[r * 128 + j];
  atomicAdd(pool + j, acc);
}

// Final Conv1d(2->1, k=1) across the two branch pools; divides by N for mean.
__global__ void final_conv_kernel(const float* __restrict__ pool,
                                  const float* __restrict__ cw,
                                  const float* __restrict__ cb,
                                  float* __restrict__ out) {
  int i = blockIdx.x * blockDim.x + threadIdx.x;
  if (i >= BATCH * D) return;
  int b = i >> 7, j = i & 127;
  const float inv = 1.0f / (float)N_NODES;
  out[i] = cw[0] * pool[(b * 2 + 0) * 128 + j] * inv +
           cw[1] * pool[(b * 2 + 1) * 128 + j] * inv + cb[0];
}

// ---------------------------------------------------------------------------
// Host
// ---------------------------------------------------------------------------
struct ParamSet { const float *Wq,*bq,*Wk,*bk,*Wv,*bv,*We,*Ws,*bs; };

extern "C" void kernel_launch(void* const* d_in, const int* in_sizes, int n_in,
                              void* d_out, int out_size, void* d_ws, size_t ws_size,
                              hipStream_t stream) {
  (void)out_size; (void)ws_size;
  auto F = [&](int i) { return (const float*)d_in[i]; };

  // ---- workspace carve-out ----
  char* ws = (char*)d_ws;
  size_t off = 0;
  auto carve = [&](size_t bytes) {
    void* p = ws + off;
    off += (bytes + 255) & ~(size_t)255;
    return p;
  };
  _Float16* Wt16 = (_Float16*)carve((size_t)16 * 16384 * sizeof(_Float16));
  float* QKVS = (float*)carve((size_t)4 * ND * sizeof(float));
  float* H    = (float*)carve((size_t)ND * sizeof(float));
  float* AGG  = (float*)carve((size_t)ND * sizeof(float));
  float* SC   = (float*)carve((size_t)N_EDGES * sizeof(float));
  float* M    = (float*)carve((size_t)N_NODES * sizeof(float));
  float* DEN  = (float*)carve((size_t)N_NODES * sizeof(float));
  float* POOL = (float*)carve((size_t)BATCH * 2 * D * sizeof(float));

  // ---- resolve parameter dicts (handles pytree-leaf or concatenated forms) ----
  ParamSet ps[4];
  const float* conv_w;
  const float* conv_b;
  if (n_in >= 42) {
    // dicts flattened to 9 leaves each; order is either jax-sorted
    // (We,Wk,Wq,Ws,Wv,bk,bq,bs,bv) or insertion (Wq,bq,Wk,bk,Wv,bv,We,Ws,bs).
    const int bases[4] = {4, 13, 22, 31};
    bool sorted_keys = (in_sizes[4] == 128);   // 'We' (128 elems) first if sorted
    for (int s = 0; s < 4; ++s) {
      int o = bases[s];
      if (sorted_keys)
        ps[s] = { F(o+2), F(o+6), F(o+1), F(o+5), F(o+4), F(o+8), F(o+0), F(o+3), F(o+7) };
      else
        ps[s] = { F(o+0), F(o+1), F(o+2), F(o+3), F(o+4), F(o+5), F(o+6), F(o+7), F(o+8) };
    }
    conv_w = F(n_in - 2);
    conv_b = F(n_in - 1);
  } else {
    // each dict is one concatenated buffer in insertion order
    for (int s = 0; s < 4; ++s) {
      const float* base = F(4 + s);
      ps[s].Wq = base;          ps[s].bq = base + 16384;
      ps[s].Wk = base + 16512;  ps[s].bk = base + 32896;
      ps[s].Wv = base + 33024;  ps[s].bv = base + 49408;
      ps[s].We = base + 49536;  ps[s].Ws = base + 49664;
      ps[s].bs = base + 66048;
    }
    conv_w = F(8);
    conv_b = F(9);
  }

  // ---- one-time prep: f16-transposed weights + pool zero ----
  WPtrs wp;
  for (int s = 0; s < 4; ++s) {
    wp.w[s * 4 + 0] = ps[s].Wq;
    wp.w[s * 4 + 1] = ps[s].Wk;
    wp.w[s * 4 + 2] = ps[s].Wv;
    wp.w[s * 4 + 3] = ps[s].Ws;
  }
  prep_weights_kernel<<<dim3(64, 16), 256, 0, stream>>>(wp, Wt16);
  zero_pool_kernel<<<4, 256, 0, stream>>>(POOL);

  const float* data  = F(0);
  const int*   adj   = (const int*)d_in[1];
  const float* etime = F(2);
  const float* esim  = F(3);

  const int gridE32 = (N_EDGES * 32 + 255) / 256;  // warp-per-edge kernels
  const int gridE1  = (N_EDGES + 255) / 256;
  const int gridND  = (ND + 255) / 256;

  for (int b = 0; b < BATCH; ++b) {
    const int* srcp = adj + (size_t)b * 2 * N_EDGES;
    const int* dstp = srcp + N_EDGES;
    const float* Xb = data + (size_t)b * ND;
    for (int br = 0; br < 2; ++br) {
      const float* ea = (br == 0 ? etime : esim) + (size_t)b * N_EDGES;
      int setA = (br == 0) ? 0 : 2;      // layer-1 params (p11 / p12)
      int setB = setA + 1;               // layer-2 params (p21 / p22)

      auto tconv = [&](const float* X, const ParamSet& p, int wset,
                       float* Hout, float* poolSlot) {
        gemm_qkvs_kernel<<<dim3(N_NODES / 16, 8), 32, 0, stream>>>(
            X, Wt16 + (size_t)wset * 4 * 16384,
            p.bq, p.bk, p.bv, p.bs, QKVS);
        init_seg_kernel<<<gridND, 256, 0, stream>>>(AGG, M, DEN);
        edge_score_kernel<<<gridE32, 256, 0, stream>>>(
            QKVS /*Q*/, QKVS + ND /*K*/, p.We, ea, srcp, dstp, SC, M);
        edge_den_kernel<<<gridE1, 256, 0, stream>>>(dstp, M, SC, DEN);
        edge_agg_kernel<<<gridE32, 256, 0, stream>>>(
            QKVS + 2 * (size_t)ND /*V*/, p.We, ea, srcp, dstp, SC, DEN, AGG);
        if (Hout)
          finalize_relu_kernel<<<gridND, 256, 0, stream>>>(
              AGG, QKVS + 3 * (size_t)ND /*skip*/, Hout);
        else
          pool_kernel<<<64, 128, 0, stream>>>(AGG, QKVS + 3 * (size_t)ND, poolSlot);
      };

      tconv(Xb, ps[setA], setA, H, nullptr);                       // layer 1 (+relu)
      tconv(H,  ps[setB], setB, nullptr, POOL + (b * 2 + br) * D); // layer 2 (+pool)
    }
  }

  final_conv_kernel<<<2, 256, 0, stream>>>(POOL, conv_w, conv_b, (float*)d_out);
}
